// EggrollTransformer_15058155339837
// MI455X (gfx1250) — matrix operand order repaired and checked by reading, not compiled, not run on hardware
//
#include <hip/hip_runtime.h>
#include <hip/hip_bf16.h>

// ---- model constants (match reference) ----
#define L_   6
#define D_   1024
#define FF_  4096
#define V_   32000
#define B_   2
#define S_   1024
#define H_   16
#define GS_  128
#define DH_  64
#define MTOK (B_ * S_)          // 2048 tokens

typedef __attribute__((ext_vector_type(16))) __bf16 v16bf;
typedef __attribute__((ext_vector_type(8)))  float  v8f;

// native f32 -> bf16 (RTNE); clang emits the gfx1250 hw cvt
__device__ __forceinline__ __bf16 f2bf(float f) { return (__bf16)f; }
__device__ __forceinline__ float bf2f(__bf16 b) {
  unsigned short s = __builtin_bit_cast(unsigned short, b);
  unsigned u = ((unsigned)s) << 16;
  return __builtin_bit_cast(float, u);
}
// pack two f32 into packed bf16x2 (low = first)
__device__ __forceinline__ unsigned pack_bf16(float lo, float hi) {
  unsigned short a = __builtin_bit_cast(unsigned short, (__bf16)lo);
  unsigned short b = __builtin_bit_cast(unsigned short, (__bf16)hi);
  return ((unsigned)b << 16) | (unsigned)a;
}

// ---- CDNA5 async global->LDS copy (ASYNCcnt path), 16B per lane ----
__device__ __forceinline__ void async_copy16(const void* gptr, const void* lptr) {
  unsigned lds = (unsigned)(size_t)lptr;   // low 32 bits = LDS byte offset
  asm volatile("global_load_async_to_lds_b128 %0, %1, off"
               :: "v"(lds), "v"(gptr) : "memory");
}
__device__ __forceinline__ void async_wait0() {
  asm volatile("s_wait_asynccnt 0" ::: "memory");
}

// ---------------------------------------------------------------------------
// Embedding: x[t,d] = emb_t[id,d]*emb_s[(id*D+d)/GS] + pos[t%S, d]
// ---------------------------------------------------------------------------
__global__ void embed_kernel(const int* __restrict__ ids,
                             const float* __restrict__ emb_t,
                             const float* __restrict__ emb_s,
                             const float* __restrict__ pos,
                             float* __restrict__ x) {
  int t = blockIdx.x;                 // 0..MTOK-1
  int s = t & (S_ - 1);
  int id = ids[t];
  const float* er = emb_t + (size_t)id * D_;
  const float* es = emb_s + (size_t)id * (D_ / GS_);
  for (int i = threadIdx.x; i < D_; i += blockDim.x)
    x[(size_t)t * D_ + i] = er[i] * es[i / GS_] + pos[(size_t)s * D_ + i];
}

// ---------------------------------------------------------------------------
// RMSNorm -> bf16 activations
// ---------------------------------------------------------------------------
__global__ __launch_bounds__(256)
void rmsnorm_kernel(const float* __restrict__ x, const float* __restrict__ w,
                    __bf16* __restrict__ h) {
  __shared__ float red[256];
  int t = blockIdx.x;
  const float* xr = x + (size_t)t * D_;
  float ss = 0.f;
  for (int i = threadIdx.x; i < D_; i += 256) { float v = xr[i]; ss += v * v; }
  red[threadIdx.x] = ss;
  __syncthreads();
  for (int o = 128; o; o >>= 1) {
    if (threadIdx.x < o) red[threadIdx.x] += red[threadIdx.x + o];
    __syncthreads();
  }
  float rs = rsqrtf(red[0] * (1.0f / D_) + 1e-6f);
  // 2 elems per thread iteration -> packed 32-bit stores
  for (int i = threadIdx.x * 2; i < D_; i += 512) {
    unsigned pk = pack_bf16(xr[i] * rs * w[i], xr[i + 1] * rs * w[i + 1]);
    *(unsigned*)&h[(size_t)t * D_ + i] = pk;
  }
}

// ---------------------------------------------------------------------------
// WMMA GEMM:  C[M,N] (+)= A_bf16[M,K] @ dequant(Wt[N,K])^T
//   - block tile 128x128, 8 waves, each wave 32x64 => 8 v_wmma per K-step
//   - K-step 32, double-buffered LDS (2 x 16 KB of the 320 KB/WGP)
//   - A tile: global_load_async_to_lds_b128 (ASYNCcnt), overlapped with WMMA
//   - B tile: ternary f32 -> regs before WMMA, hw-cvt dequant -> LDS after
// All of M=2048, N in {1024,4096,32000}, K in {1024,4096} divide tiles exactly.
// ---------------------------------------------------------------------------
__global__ __launch_bounds__(256)
void gemm_tern_bf16(const __bf16* __restrict__ A, const float* __restrict__ Wt,
                    const float* __restrict__ Ws, float* __restrict__ C,
                    int N, int K, int beta) {
  __shared__ __bf16 lsA[2][128 * 32];   // 2 x 8 KB
  __shared__ __bf16 lsB[2][128 * 32];   // 2 x 8 KB
  const int tid   = threadIdx.x;
  const int lane  = tid & 31;
  const int wave  = tid >> 5;
  const int waveM = wave >> 1;          // 0..3  (M offset = waveM*32)
  const int waveN = wave & 1;           // 0..1  (N offset = waveN*64)
  const int bM = blockIdx.y * 128;
  const int bN = blockIdx.x * 128;
  const int kg = K / GS_;
  const int l15 = lane & 15;
  const int kbA = (lane >> 4) * 8;      // ISA 16-bit A 16x32 frag K-base
  const int kbB = (lane >> 4) * 8;      // ISA B 32x16 frag K-base

  v8f zero = {};
  v8f acc[2][4];
  #pragma unroll
  for (int mi = 0; mi < 2; mi++)
    #pragma unroll
    for (int ni = 0; ni < 4; ni++) acc[mi][ni] = zero;

  // ---- prologue: stage K-tile 0 into buffer 0 ----
  #pragma unroll
  for (int c = 0; c < 2; c++) {         // A: 512 x 16B chunks / 256 threads
    int ch = tid + c * 256, row = ch >> 2, col = (ch & 3) * 8;
    async_copy16(A + (size_t)(bM + row) * K + col, &lsA[0][row * 32 + col]);
  }
  #pragma unroll
  for (int c = 0; c < 4; c++) {         // B: 1024 float4 chunks / 256 threads
    int ch = tid + c * 256, row = ch >> 3, col = (ch & 7) * 4;
    int n = bN + row;
    float4 wv = *(const float4*)(Wt + (size_t)n * K + col);
    float sc = Ws[(size_t)n * kg + (unsigned)col / GS_];
    uint2 pk = { pack_bf16(wv.x * sc, wv.y * sc),
                 pack_bf16(wv.z * sc, wv.w * sc) };
    *(uint2*)&lsB[0][row * 32 + col] = pk;
  }
  async_wait0();
  __syncthreads();

  const int nk = K / 32;
  for (int it = 0; it < nk; ++it) {
    const int cur = it & 1, nxt = cur ^ 1;
    const int k1 = (it + 1) * 32;
    const bool pre = (it + 1 < nk);

    float4 wreg[4];
    float  sreg[4];
    if (pre) {
      // issue next A tile via async engine (lands in LDS during WMMAs)
      #pragma unroll
      for (int c = 0; c < 2; c++) {
        int ch = tid + c * 256, row = ch >> 2, col = (ch & 3) * 8;
        async_copy16(A + (size_t)(bM + row) * K + k1 + col,
                     &lsA[nxt][row * 32 + col]);
      }
      // issue next B tile global loads into registers
      #pragma unroll
      for (int c = 0; c < 4; c++) {
        int ch = tid + c * 256, row = ch >> 3, col = (ch & 7) * 4;
        int n = bN + row;
        wreg[c] = *(const float4*)(Wt + (size_t)n * K + k1 + col);
        sreg[c] = Ws[(size_t)n * kg + (unsigned)(k1 + col) / GS_];
      }
    }

    // ---- fragments from current buffer (wave32 WMMA layouts) ----
    v16bf af[2], bfr[4];
    #pragma unroll
    for (int mi = 0; mi < 2; mi++) {
      const __bf16* ap = &lsA[cur][(waveM * 32 + mi * 16 + l15) * 32];
      #pragma unroll
      for (int i = 0; i < 16; i++) af[mi][i] = ap[kbA + (i < 8 ? i : i + 8)];
    }
    #pragma unroll
    for (int ni = 0; ni < 4; ni++) {
      const __bf16* bp = &lsB[cur][(waveN * 64 + ni * 16 + l15) * 32 + kbB];
      #pragma unroll
      for (int i = 0; i < 16; i++) bfr[ni][i] = bp[i < 8 ? i : i + 8];
    }
    #pragma unroll
    for (int mi = 0; mi < 2; mi++)
      #pragma unroll
      for (int ni = 0; ni < 4; ni++)
        acc[mi][ni] = __builtin_amdgcn_wmma_f32_16x16x32_bf16(
            false, af[mi], false, bfr[ni], (short)0, acc[mi][ni],
            false, false);

    if (pre) {
      // hw-cvt dequant of staged B registers into the other LDS buffer
      #pragma unroll
      for (int c = 0; c < 4; c++) {
        int ch = tid + c * 256, row = ch >> 3, col = (ch & 7) * 4;
        uint2 pk = { pack_bf16(wreg[c].x * sreg[c], wreg[c].y * sreg[c]),
                     pack_bf16(wreg[c].z * sreg[c], wreg[c].w * sreg[c]) };
        *(uint2*)&lsB[nxt][row * 32 + col] = pk;
      }
    }
    async_wait0();
    __syncthreads();
  }

  // ---- writeback (C layout: lane halves hold M = 8*(l>>4)+vgpr, N = l&15) --
  #pragma unroll
  for (int mi = 0; mi < 2; mi++)
    #pragma unroll
    for (int ni = 0; ni < 4; ni++) {
      int n = bN + waveN * 64 + ni * 16 + l15;
      int mbase = bM + waveM * 32 + mi * 16 + ((lane >> 4) * 8);
      #pragma unroll
      for (int j = 0; j < 8; j++) {
        size_t idx = (size_t)(mbase + j) * N + n;
        float vv = acc[mi][ni][j];
        C[idx] = beta ? (C[idx] + vv) : vv;
      }
    }
}

// ---------------------------------------------------------------------------
// Attention: one wave per query, online softmax over keys (lane-partitioned),
// + alpha * normed-input head mix; output bf16 ready for O-proj GEMM.
// q/k/v: f32 [token, h*DH+dh]
// ---------------------------------------------------------------------------
__global__ __launch_bounds__(256)
void attn_kernel(const float* __restrict__ q, const float* __restrict__ k,
                 const float* __restrict__ v, const __bf16* __restrict__ hn,
                 const float* __restrict__ alpha_l, __bf16* __restrict__ out) {
  const int lane = threadIdx.x & 31;
  const int wave = threadIdx.x >> 5;
  const int bh = blockIdx.x;                  // 0..B*H-1
  const int b = bh / H_, hd = bh % H_;
  const int qi = blockIdx.y * 8 + wave;       // 0..S-1
  const size_t base = (size_t)(b * S_) * D_ + hd * DH_;
  const float* qp = q + base + (size_t)qi * D_;

  float qreg[DH_];
  #pragma unroll
  for (int d = 0; d < DH_; d++) qreg[d] = qp[d];

  float m = -__builtin_inff(), l = 0.f;
  float o[DH_];
  #pragma unroll
  for (int d = 0; d < DH_; d++) o[d] = 0.f;

  for (int j = lane; j <= qi; j += 32) {
    const float* kp = k + base + (size_t)j * D_;
    float s = 0.f;
    #pragma unroll
    for (int d = 0; d < DH_; d++) s += qreg[d] * kp[d];
    s *= 0.125f;                              // DH^-0.5
    float mnew = fmaxf(m, s);
    float corr = __expf(m - mnew);
    float p = __expf(s - mnew);
    l = l * corr + p;
    const float* vp = v + base + (size_t)j * D_;
    #pragma unroll
    for (int d = 0; d < DH_; d++) o[d] = o[d] * corr + p * vp[d];
    m = mnew;
  }

  // combine lane-local softmax states across the wave
  float mg = m;
  #pragma unroll
  for (int off = 16; off; off >>= 1) mg = fmaxf(mg, __shfl_xor(mg, off, 32));
  float c = __expf(m - mg);                   // lanes with no keys -> 0
  l *= c;
  #pragma unroll
  for (int d = 0; d < DH_; d++) o[d] *= c;
  #pragma unroll
  for (int off = 16; off; off >>= 1) {
    l += __shfl_xor(l, off, 32);
    #pragma unroll
    for (int d = 0; d < DH_; d++) o[d] += __shfl_xor(o[d], off, 32);
  }

  float inv = 1.f / l;
  float al = alpha_l[hd];
  size_t ob = base + (size_t)qi * D_;
  #pragma unroll
  for (int d = 0; d < DH_; d += 2) {
    if ((d >> 1) == lane) {                   // constant index into o[]
      float lo = o[d]     * inv + al * bf2f(hn[ob + d]);
      float hi = o[d + 1] * inv + al * bf2f(hn[ob + d + 1]);
      *(unsigned*)&out[ob + d] = pack_bf16(lo, hi);
    }
  }
}

// ---------------------------------------------------------------------------
// act = bf16( silu(g) * u ), packed pair per thread
// ---------------------------------------------------------------------------
__global__ void silu_mul_kernel(const float* __restrict__ g,
                                const float* __restrict__ u,
                                __bf16* __restrict__ a, int n) {
  int i = (blockIdx.x * blockDim.x + threadIdx.x) * 2;
  if (i < n) {
    float x0 = g[i],     s0 = x0 / (1.f + __expf(-x0));
    float x1 = g[i + 1], s1 = x1 / (1.f + __expf(-x1));
    *(unsigned*)&a[i] = pack_bf16(s0 * u[i], s1 * u[i + 1]);
  }
}

// ---------------------------------------------------------------------------
extern "C" void kernel_launch(void* const* d_in, const int* in_sizes, int n_in,
                              void* d_out, int out_size, void* d_ws, size_t ws_size,
                              hipStream_t stream) {
  const int*   ids    = (const int*)  d_in[0];
  const float* emb_t  = (const float*)d_in[1];
  const float* emb_s  = (const float*)d_in[2];
  const float* pos    = (const float*)d_in[3];
  const float* q_t    = (const float*)d_in[4];
  const float* q_s    = (const float*)d_in[5];
  const float* k_t    = (const float*)d_in[6];
  const float* k_s    = (const float*)d_in[7];
  const float* v_t    = (const float*)d_in[8];
  const float* v_s    = (const float*)d_in[9];
  const float* o_t    = (const float*)d_in[10];
  const float* o_s    = (const float*)d_in[11];
  const float* g_t    = (const float*)d_in[12];
  const float* g_s    = (const float*)d_in[13];
  const float* u_t    = (const float*)d_in[14];
  const float* u_s    = (const float*)d_in[15];
  const float* dw_t   = (const float*)d_in[16];
  const float* dw_s   = (const float*)d_in[17];
  const float* alpha  = (const float*)d_in[18];
  const float* na_w   = (const float*)d_in[19];
  const float* nm_w   = (const float*)d_in[20];
  const float* nf_w   = (const float*)d_in[21];
  const float* head_t = (const float*)d_in[22];
  const float* head_s = (const float*)d_in[23];
  float* logits = (float*)d_out;

  // ---- workspace carve-out (all offsets 16B+ aligned by construction) ----
  char* p = (char*)d_ws;
  float*  x    = (float*) p; p += (size_t)MTOK * D_  * 4;   // 8 MB
  __bf16* h    = (__bf16*)p; p += (size_t)MTOK * D_  * 2;   // 4 MB
  float*  qb   = (float*) p; p += (size_t)MTOK * D_  * 4;
  float*  kb   = (float*) p; p += (size_t)MTOK * D_  * 4;
  float*  vb   = (float*) p; p += (size_t)MTOK * D_  * 4;
  __bf16* att  = (__bf16*)p; p += (size_t)MTOK * D_  * 2;
  float*  gbuf = (float*) p; p += (size_t)MTOK * FF_ * 4;   // 32 MB
  float*  ubuf = (float*) p; p += (size_t)MTOK * FF_ * 4;   // 32 MB
  __bf16* act  = (__bf16*)p; p += (size_t)MTOK * FF_ * 2;   // 16 MB

  const dim3 blk(256);
  const dim3 gD (D_  / 128, MTOK / 128);  // N=1024 GEMMs
  const dim3 gFF(FF_ / 128, MTOK / 128);  // N=4096 GEMMs
  const dim3 gV (V_  / 128, MTOK / 128);  // head GEMM
  const dim3 gAtt(B_ * H_, S_ / 8);

  embed_kernel<<<MTOK, blk, 0, stream>>>(ids, emb_t, emb_s, pos, x);

  const size_t WQ = (size_t)D_ * D_,  SQ = WQ / GS_;
  const size_t WF = (size_t)FF_ * D_, SF = WF / GS_;

  for (int l = 0; l < L_; l++) {
    rmsnorm_kernel<<<MTOK, blk, 0, stream>>>(x, na_w + (size_t)l * D_, h);

    gemm_tern_bf16<<<gD, blk, 0, stream>>>(h, q_t + l * WQ, q_s + l * SQ, qb, D_, D_, 0);
    gemm_tern_bf16<<<gD, blk, 0, stream>>>(h, k_t + l * WQ, k_s + l * SQ, kb, D_, D_, 0);
    gemm_tern_bf16<<<gD, blk, 0, stream>>>(h, v_t + l * WQ, v_s + l * SQ, vb, D_, D_, 0);

    attn_kernel<<<gAtt, blk, 0, stream>>>(qb, kb, vb, h, alpha + (size_t)l * H_, att);

    gemm_tern_bf16<<<gD, blk, 0, stream>>>(att, o_t + l * WQ, o_s + l * SQ, x, D_, D_, 1);

    rmsnorm_kernel<<<MTOK, blk, 0, stream>>>(x, nm_w + (size_t)l * D_, h);

    gemm_tern_bf16<<<gFF, blk, 0, stream>>>(h, g_t + l * WF, g_s + l * SF, gbuf, FF_, D_, 0);
    gemm_tern_bf16<<<gFF, blk, 0, stream>>>(h, u_t + l * WF, u_s + l * SF, ubuf, FF_, D_, 0);

    int nact = MTOK * FF_;
    silu_mul_kernel<<<(nact / 2 + 255) / 256, blk, 0, stream>>>(gbuf, ubuf, act, nact);

    gemm_tern_bf16<<<gD, blk, 0, stream>>>(act, dw_t + l * WQ * 4, dw_s + l * SF, x, D_, FF_, 1);
  }

  rmsnorm_kernel<<<MTOK, blk, 0, stream>>>(x, nf_w, h);
  gemm_tern_bf16<<<gV, blk, 0, stream>>>(h, head_t, head_s, logits, V_, D_, 0);
}
